// SMPE2_Encoder_19988777796329
// MI455X (gfx1250) — compile-verified, matrix operand order checked by reading, not run on hardware
//
#include <hip/hip_runtime.h>
#include <hip/hip_bf16.h>

#define OBS    128
#define ACTD   32
#define HID    64
#define DIN    160      // OBS + ACTD
#define TMAX   8
#define CENTER 7

typedef __attribute__((ext_vector_type(16))) _Float16 v16h;
typedef __attribute__((ext_vector_type(8)))  float    v8f;

// ---------------------------------------------------------------- WMMA helpers
__device__ __forceinline__ v8f wmma_f16(v16h a, v16h b, v8f c) {
  // D = A(16x32 f16) * B(32x16 f16) + C(16x16 f32)
  return __builtin_amdgcn_wmma_f32_16x16x32_f16(
      /*neg_a=*/false, a, /*neg_b=*/false, b,
      /*c_mod=*/(short)0, c, /*reuse_a=*/false, /*reuse_b=*/false);
}

// A fragment: 16 rows x 32 K from row-major f16 LDS [16][stride].
// ISA layout: lanes 0-15 -> M=lane, K {0..7,16..23}; lanes 16-31 -> K {8..15,24..31}.
__device__ __forceinline__ v16h frag_a_lds(const _Float16* base, int stride,
                                           int kbase, int lane) {
  const int m = lane & 15, half = lane >> 4;
  const _Float16* p = base + m * stride + kbase + half * 8;
  v16h a;
#pragma unroll
  for (int i = 0; i < 16; ++i) a[i] = p[((i >> 3) << 4) + (i & 7)];
  return a;
}

// B fragment for D = X @ W^T : B[k][n] = W[n][k], W row-major f16 [N][K] in LDS.
__device__ __forceinline__ v16h frag_bT_lds(const _Float16* W, int K,
                                            int nbase, int kbase, int lane) {
  const int n = nbase + (lane & 15), half = lane >> 4;
  const _Float16* p = W + n * K + kbase + half * 8;
  v16h b;
#pragma unroll
  for (int i = 0; i < 16; ++i) b[i] = p[((i >> 3) << 4) + (i & 7)];
  return b;
}

// Same but sourcing f32 weights straight from global (small head weights).
__device__ __forceinline__ v16h frag_bT_g32(const float* W, int K,
                                            int nbase, int kbase, int lane) {
  const int n = nbase + (lane & 15), half = lane >> 4;
  const float* p = W + n * K + kbase + half * 8;
  v16h b;
#pragma unroll
  for (int i = 0; i < 16; ++i) b[i] = (_Float16)p[((i >> 3) << 4) + (i & 7)];
  return b;
}

__device__ __forceinline__ float wred_sum(float v) {
#pragma unroll
  for (int off = 16; off > 0; off >>= 1) v += __shfl_xor(v, off, 32);
  return v;
}
__device__ __forceinline__ float wred_max(float v) {
#pragma unroll
  for (int off = 16; off > 0; off >>= 1) v = fmaxf(v, __shfl_xor(v, off, 32));
  return v;
}

// ------------------------------------------------- Kernel 1: features + window
__global__ void __launch_bounds__(128)
smpe2_feat_kernel(const float* __restrict__ obs, const float* __restrict__ act,
                  const float* __restrict__ ln_g, const float* __restrict__ ln_b,
                  const float* __restrict__ dw1, const float* __restrict__ db1,
                  const float* __restrict__ dw2, const float* __restrict__ db2,
                  const float* __restrict__ dw3, const float* __restrict__ db3,
                  int* __restrict__ lenF, int* __restrict__ lenB, int B) {
  const int lane = threadIdx.x & 31;
  const int row  = blockIdx.x * 4 + (threadIdx.x >> 5);
  if (row >= B) return;

  const float* obs_t = obs + ((size_t)row * 15 + CENTER) * OBS;
  float o[4];
#pragma unroll
  for (int j = 0; j < 4; ++j) o[j] = obs_t[lane + 32 * j];

  // --- entropy of softmax(obs_t)
  float mx = fmaxf(fmaxf(o[0], o[1]), fmaxf(o[2], o[3]));
  mx = wred_max(mx);
  float s = 0.f;
#pragma unroll
  for (int j = 0; j < 4; ++j) s += __expf(o[j] - mx);
  s = wred_sum(s);
  float ent = 0.f;
#pragma unroll
  for (int j = 0; j < 4; ++j) {
    float p = __expf(o[j] - mx) / s;
    ent -= p * __logf(p + 1e-8f);
  }
  ent = wred_sum(ent);

  // --- rate of change: mean of 3 adjacent-frame L2 norms
  float roc = 0.f;
#pragma unroll
  for (int k = 0; k < 3; ++k) {
    const float* a0 = obs + ((size_t)row * 15 + (CENTER - k)) * OBS;
    const float* a1 = obs + ((size_t)row * 15 + (CENTER - k - 1)) * OBS;
    float d = 0.f;
#pragma unroll
    for (int j = 0; j < 4; ++j) {
      float t = a0[lane + 32 * j] - a1[lane + 32 * j];
      d += t * t;
    }
    roc += sqrtf(wred_sum(d));
  }
  roc *= (1.f / 3.f);

  // --- correlation of obs_t with zero-padded act_prev (both length 128)
  const float* ap = act + ((size_t)row * 15 + (CENTER - 1)) * ACTD;
  float av = ap[lane];  // only j==0 slot (i<32) carries act data
  float mean_o = wred_sum(o[0] + o[1] + o[2] + o[3]) * (1.f / 128.f);
  float mean_a = wred_sum(av) * (1.f / 128.f);
  float dot = 0.f, no = 0.f, na = 0.f;
#pragma unroll
  for (int j = 0; j < 4; ++j) {
    float oc = o[j] - mean_o;
    float ac = (j == 0 ? av : 0.f) - mean_a;
    dot += oc * ac; no += oc * oc; na += ac * ac;
  }
  dot = wred_sum(dot); no = wred_sum(no); na = wred_sum(na);
  float corr = dot / (sqrtf(no) * sqrtf(na) + 1e-8f);

  // --- layernorm over the 3 features
  float f0 = ent, f1 = roc, f2 = corr;
  float mean = (f0 + f1 + f2) * (1.f / 3.f);
  float v0 = f0 - mean, v1 = f1 - mean, v2 = f2 - mean;
  float var = (v0 * v0 + v1 * v1 + v2 * v2) * (1.f / 3.f);
  float inv = rsqrtf(var + 1e-5f);
  float fn[3];
  fn[0] = v0 * inv * ln_g[0] + ln_b[0];
  fn[1] = v1 * inv * ln_g[1] + ln_b[1];
  fn[2] = v2 * inv * ln_g[2] + ln_b[2];

  // --- tiny MLP 3->16->8->14 + argmax (redundant per lane; lane 0 writes)
  float h1[16];
#pragma unroll
  for (int i = 0; i < 16; ++i) {
    float acc = db1[i];
#pragma unroll
    for (int j = 0; j < 3; ++j) acc += dw1[i * 3 + j] * fn[j];
    h1[i] = fmaxf(acc, 0.f);
  }
  float h2[8];
#pragma unroll
  for (int i = 0; i < 8; ++i) {
    float acc = db2[i];
#pragma unroll
    for (int j = 0; j < 16; ++j) acc += dw2[i * 16 + j] * h1[j];
    h2[i] = fmaxf(acc, 0.f);
  }
  float best = -1e30f; int bi = 0;
#pragma unroll
  for (int i = 0; i < 14; ++i) {
    float acc = db3[i];
#pragma unroll
    for (int j = 0; j < 8; ++j) acc += dw3[i * 8 + j] * h2[j];
    if (acc > best) { best = acc; bi = i; }   // first-max like jnp.argmax
  }
  const int wl = bi + 2;
  if (lane == 0) {
    lenF[row] = (wl - 1) / 2 + 1;   // in [1,8]
    lenB[row] = wl / 2 + 1;         // in [2,8]
  }
}

// --------------------------------------------------- GRU scan (one direction)
// Wave wv owns hidden columns [16*wv, 16*wv+16) for ALL three gates, so the
// nonlinear update stays in registers; h is exchanged via a 2KB LDS tile.
__device__ __forceinline__ void gru_scan(const _Float16* xseq, const _Float16* WihL,
                                         const v16h* whh, const float* bih,
                                         const float* bhh, const int* lenS,
                                         _Float16* hbuf, _Float16* comb,
                                         int coloff, int lane, int wv) {
  const int nb = wv * 16;
  const int ncol = nb + (lane & 15);
  const int half = lane >> 4;
  const float bi_r = bih[ncol],       bh_r = bhh[ncol];
  const float bi_z = bih[64 + ncol],  bh_z = bhh[64 + ncol];
  const float bi_n = bih[128 + ncol], bh_n = bhh[128 + ncol];
  int lenv[8];
#pragma unroll
  for (int r = 0; r < 8; ++r) lenv[r] = lenS[half * 8 + r];

  v8f h = {};
  for (int t = 0; t < TMAX; ++t) {
    // gh = h_{t-1} @ Whh^T for this wave's 3 gate tiles (Whh frags in regs)
    v16h ah0 = frag_a_lds(hbuf, HID, 0, lane);
    v16h ah1 = frag_a_lds(hbuf, HID, 32, lane);
    __syncthreads();  // everyone has read hbuf; safe to overwrite below
    v8f gr = {}, gz = {}, gn = {};
    gr = wmma_f16(ah0, whh[0], gr); gr = wmma_f16(ah1, whh[1], gr);
    gz = wmma_f16(ah0, whh[2], gz); gz = wmma_f16(ah1, whh[3], gz);
    gn = wmma_f16(ah0, whh[4], gn); gn = wmma_f16(ah1, whh[5], gn);

    // gi = x_t @ Wih^T, K=160 in 5 chunks
    v8f ir = {}, iz = {}, inn = {};
    const _Float16* xa = xseq + t * 16 * DIN;
#pragma unroll
    for (int kc = 0; kc < 5; ++kc) {
      v16h a = frag_a_lds(xa, DIN, kc * 32, lane);
      ir  = wmma_f16(a, frag_bT_lds(WihL, DIN, nb,       kc * 32, lane), ir);
      iz  = wmma_f16(a, frag_bT_lds(WihL, DIN, 64 + nb,  kc * 32, lane), iz);
      inn = wmma_f16(a, frag_bT_lds(WihL, DIN, 128 + nb, kc * 32, lane), inn);
    }

    // gate nonlinearity + masked state update, entirely in-register
#pragma unroll
    for (int r = 0; r < 8; ++r) {
      float rg = 1.f / (1.f + __expf(-(ir[r] + bi_r + gr[r] + bh_r)));
      float zg = 1.f / (1.f + __expf(-(iz[r] + bi_z + gz[r] + bh_z)));
      float nn = tanhf(inn[r] + bi_n + rg * (gn[r] + bh_n));
      float hn = (1.f - zg) * nn + zg * h[r];
      h[r] = (t < lenv[r]) ? hn : h[r];
    }
#pragma unroll
    for (int r = 0; r < 8; ++r)
      hbuf[(half * 8 + r) * HID + ncol] = (_Float16)h[r];
    __syncthreads();
  }
#pragma unroll
  for (int r = 0; r < 8; ++r)
    comb[(half * 8 + r) * (2 * HID) + coloff + ncol] = (_Float16)h[r];
}

// --------------------------------------- Kernel 2: bidirectional GRU + heads
__global__ void __launch_bounds__(128)
smpe2_gru_kernel(const float* __restrict__ obs, const float* __restrict__ act,
                 const float* __restrict__ Wih_f, const float* __restrict__ Whh_f,
                 const float* __restrict__ bih_f, const float* __restrict__ bhh_f,
                 const float* __restrict__ Wih_b, const float* __restrict__ Whh_b,
                 const float* __restrict__ bih_b, const float* __restrict__ bhh_b,
                 const float* __restrict__ mw1, const float* __restrict__ mb1,
                 const float* __restrict__ mw2, const float* __restrict__ mb2,
                 const float* __restrict__ mu_w, const float* __restrict__ mu_b,
                 const float* __restrict__ lv_w, const float* __restrict__ lv_b,
                 const int* __restrict__ lenFg, const int* __restrict__ lenBg,
                 float* __restrict__ out, int B) {
  extern __shared__ char smem_raw[];
  _Float16* xseq = (_Float16*)smem_raw;        // 8*16*160  = 20480 halfs
  _Float16* WihL = xseq + TMAX * 16 * DIN;     // 192*160   = 30720
  _Float16* hbuf = WihL + 192 * DIN;           // 16*64     = 1024
  _Float16* comb = hbuf + 16 * HID;            // 16*128    = 2048
  _Float16* h1b  = comb + 16 * 2 * HID;        // 16*64     = 1024
  _Float16* zeb  = h1b + 16 * HID;             // 16*64     = 1024
  int*      lenS = (int*)(zeb + 16 * HID);     // 16 ints

  const int tid = threadIdx.x;
  const int lane = tid & 31, wv = tid >> 5;
  const int row0 = blockIdx.x * 16;

  for (int dir = 0; dir < 2; ++dir) {
    const int* lenG = dir ? lenBg : lenFg;
    if (tid < 16) lenS[tid] = lenG[row0 + tid];
    for (int e = tid; e < 16 * HID; e += 128) hbuf[e] = (_Float16)0.f;
    __syncthreads();

    // stage gathered sequence [t][m][k] as f16 (window mask redundant here:
    // all valid timesteps lie inside the window; invalid ones are h-masked)
    for (int e = tid; e < TMAX * 16 * DIN; e += 128) {
      int t = e / (16 * DIN);
      int rem = e - t * (16 * DIN);
      int m = rem / DIN;
      int k = rem - m * DIN;
      int len = lenS[m];
      int tt = dir ? (CENTER - 1 + len - t) : (CENTER + 1 - len + t);  // in [0,14]
      size_t rr = (size_t)(row0 + m) * 15 + tt;
      float v = (k < OBS) ? obs[rr * OBS + k] : act[rr * ACTD + (k - OBS)];
      xseq[e] = (_Float16)v;
    }
    // stage Wih (192x160) as f16 into LDS
    const float* Wih = dir ? Wih_b : Wih_f;
    for (int e = tid; e < 192 * DIN; e += 128) WihL[e] = (_Float16)Wih[e];
    // Whh fragments register-resident (6 x v16h per wave)
    const float* Whh = dir ? Whh_b : Whh_f;
    v16h whh[6];
#pragma unroll
    for (int g = 0; g < 3; ++g)
#pragma unroll
      for (int kc = 0; kc < 2; ++kc)
        whh[g * 2 + kc] = frag_bT_g32(Whh, HID, g * 64 + wv * 16, kc * 32, lane);
    __syncthreads();

    gru_scan(xseq, WihL, whh, dir ? bih_b : bih_f, dir ? bhh_b : bhh_f,
             lenS, hbuf, comb, dir ? HID : 0, lane, wv);
    __syncthreads();
  }

  // ---- heads: comb(16x128) -> relu L1 -> L2 -> {mu, logvar}
  const int nb = wv * 16;
  const int ncol = nb + (lane & 15);
  const int half = lane >> 4;

  v8f a1 = {};
#pragma unroll
  for (int kc = 0; kc < 4; ++kc) {
    v16h a = frag_a_lds(comb, 2 * HID, kc * 32, lane);
    a1 = wmma_f16(a, frag_bT_g32(mw1, 2 * HID, nb, kc * 32, lane), a1);
  }
  {
    float b1 = mb1[ncol];
#pragma unroll
    for (int r = 0; r < 8; ++r)
      h1b[(half * 8 + r) * HID + ncol] = (_Float16)fmaxf(a1[r] + b1, 0.f);
  }
  __syncthreads();

  v8f a2 = {};
#pragma unroll
  for (int kc = 0; kc < 2; ++kc) {
    v16h a = frag_a_lds(h1b, HID, kc * 32, lane);
    a2 = wmma_f16(a, frag_bT_g32(mw2, HID, nb, kc * 32, lane), a2);
  }
  {
    float b2 = mb2[ncol];
#pragma unroll
    for (int r = 0; r < 8; ++r)
      zeb[(half * 8 + r) * HID + ncol] = (_Float16)(a2[r] + b2);
  }
  __syncthreads();

  v8f am = {}, al = {};
#pragma unroll
  for (int kc = 0; kc < 2; ++kc) {
    v16h a = frag_a_lds(zeb, HID, kc * 32, lane);
    am = wmma_f16(a, frag_bT_g32(mu_w, HID, nb, kc * 32, lane), am);
    al = wmma_f16(a, frag_bT_g32(lv_w, HID, nb, kc * 32, lane), al);
  }
  const float bm = mu_b[ncol], bl = lv_b[ncol];
  const size_t NB = (size_t)B * HID;
#pragma unroll
  for (int r = 0; r < 8; ++r) {
    size_t o = (size_t)(row0 + half * 8 + r) * HID + ncol;
    float mu = am[r] + bm;
    float sg = __expf(0.5f * (al[r] + bl));
    out[o] = mu;            // z
    out[NB + o] = mu;       // mu
    out[2 * NB + o] = sg;   // sigma
  }
}

// ------------------------------------------------------------------- launcher
extern "C" void kernel_launch(void* const* d_in, const int* in_sizes, int n_in,
                              void* d_out, int out_size, void* d_ws, size_t ws_size,
                              hipStream_t stream) {
  const float* obs   = (const float*)d_in[0];
  const float* act   = (const float*)d_in[1];
  const float* ln_g  = (const float*)d_in[2];
  const float* ln_b  = (const float*)d_in[3];
  const float* dw1   = (const float*)d_in[4];
  const float* db1   = (const float*)d_in[5];
  const float* dw2   = (const float*)d_in[6];
  const float* db2   = (const float*)d_in[7];
  const float* dw3   = (const float*)d_in[8];
  const float* db3   = (const float*)d_in[9];
  const float* Wih_f = (const float*)d_in[10];
  const float* Whh_f = (const float*)d_in[11];
  const float* bih_f = (const float*)d_in[12];
  const float* bhh_f = (const float*)d_in[13];
  const float* Wih_b = (const float*)d_in[14];
  const float* Whh_b = (const float*)d_in[15];
  const float* bih_b = (const float*)d_in[16];
  const float* bhh_b = (const float*)d_in[17];
  const float* mw1   = (const float*)d_in[18];
  const float* mb1   = (const float*)d_in[19];
  const float* mw2   = (const float*)d_in[20];
  const float* mb2   = (const float*)d_in[21];
  const float* mu_w  = (const float*)d_in[22];
  const float* mu_b  = (const float*)d_in[23];
  const float* lv_w  = (const float*)d_in[24];
  const float* lv_b  = (const float*)d_in[25];

  const int B = in_sizes[0] / (15 * OBS);
  int* lenF = (int*)d_ws;
  int* lenB = lenF + B;
  float* out = (float*)d_out;

  smpe2_feat_kernel<<<(B + 3) / 4, 128, 0, stream>>>(
      obs, act, ln_g, ln_b, dw1, db1, dw2, db2, dw3, db3, lenF, lenB, B);

  const size_t lds_bytes =
      (size_t)(TMAX * 16 * DIN + 192 * DIN + 16 * HID + 16 * 2 * HID +
               16 * HID + 16 * HID) * sizeof(_Float16) + 16 * sizeof(int);
  smpe2_gru_kernel<<<B / 16, 128, lds_bytes, stream>>>(
      obs, act, Wih_f, Whh_f, bih_f, bhh_f, Wih_b, Whh_b, bih_b, bhh_b,
      mw1, mb1, mw2, mb2, mu_w, mu_b, lv_w, lv_b, lenF, lenB, out, B);
}
